// LSTM_7473243095127
// MI455X (gfx1250) — compile-verified, matrix operand order checked by reading, not run on hardware
//
#include <hip/hip_runtime.h>

typedef __attribute__((ext_vector_type(16))) __bf16 v16bf;
typedef __attribute__((ext_vector_type(2)))  __bf16 v2bf;
typedef __attribute__((ext_vector_type(8)))  float  v8f;
typedef __attribute__((ext_vector_type(2)))  float  v2f;

#define BATCH 4096
#define UNITS 2048
#define INPD  2048
#define KTOT  4096   // 2048 (x @ W^T) + 2048 (pre_h @ U^T)

#define BM 128       // batch rows per block
#define BN 64        // units per block (per gate)
#define BK 64        // K elements staged per iteration
#define NSTAGES (KTOT / BK)
#define NTHREADS 512 // 16 waves: 4 (M) x 4 (N)

#define A_ELEMS (BM * BK)          // 8192 bf16 per A buffer
#define B_ELEMS (4 * BN * BK)      // 16384 bf16 per B buffer
#define B_BASE  (2 * A_ELEMS)      // B buffers start after both A buffers
#define LDS_BYTES (2 * (A_ELEMS + B_ELEMS) * 2)   // 98304 = 96 KB

#define A_LD 4   // float4 loads per thread for A stage (2048/512)
#define B_LD 8   // float4 loads per thread for B stage (4096/512)

__device__ __forceinline__ float fsigmoid(float x) {
  return 1.0f / (1.0f + __expf(-x));
}
__device__ __forceinline__ float ftanh(float x) {
  return 2.0f / (1.0f + __expf(-2.0f * x)) - 1.0f;
}

// two fp32 -> packed bf16x2 in one hardware convert (v_cvt_pk_bf16_f32)
__device__ __forceinline__ unsigned int cvt_pk2(float a, float b) {
  v2f f = {a, b};
  union { v2bf h; unsigned int u; } r;
  r.h = __builtin_convertvector(f, v2bf);
  return r.u;
}
__device__ __forceinline__ uint2 pk_bf16x4(float4 v) {
  uint2 r;
  r.x = cvt_pk2(v.x, v.y);
  r.y = cvt_pk2(v.z, v.w);
  return r;
}

union BFrag { v16bf v; uint4 q[2]; };

// issue global loads for one K stage into registers (no waits here)
__device__ __forceinline__ void load_stage(
    const float* __restrict__ x, const float* __restrict__ pre_h,
    const float* __restrict__ W, const float* __restrict__ U,
    int k0, int blockM, int blockN, int tid,
    float4 sa[A_LD], float4 sb[B_LD])
{
  const bool firstHalf = (k0 < INPD);
  const float* Asrc = firstHalf ? x : pre_h;
  const float* Bsrc = firstHalf ? W : U;
  const int kc = firstHalf ? k0 : (k0 - INPD);

#pragma unroll
  for (int i = 0; i < A_LD; ++i) {
    const int idx = tid + i * NTHREADS;
    const int row = idx >> 4;       // BK/4 = 16 float4 per row
    const int c4  = idx & 15;
    sa[i] = *(const float4*)(Asrc + (size_t)(blockM + row) * 2048 + kc + c4 * 4);
  }
#pragma unroll
  for (int i = 0; i < B_LD; ++i) {
    const int idx = tid + i * NTHREADS;
    const int g   = idx >> 10;            // BN*BK/4 = 1024 per gate
    const int row = (idx >> 4) & (BN - 1);
    const int c4  = idx & 15;
    sb[i] = *(const float4*)(Bsrc + (size_t)g * UNITS * 2048
                             + (size_t)(blockN + row) * 2048 + kc + c4 * 4);
  }
}

// convert staged registers to bf16 and store into an LDS buffer (by offset)
__device__ __forceinline__ void store_stage(
    unsigned short* smem, int aOff, int bOff, int tid,
    const float4 sa[A_LD], const float4 sb[B_LD])
{
#pragma unroll
  for (int i = 0; i < A_LD; ++i) {
    const int idx = tid + i * NTHREADS;
    const int row = idx >> 4;
    const int c4  = idx & 15;
    *(uint2*)(&smem[aOff + row * BK + c4 * 4]) = pk_bf16x4(sa[i]);
  }
#pragma unroll
  for (int i = 0; i < B_LD; ++i) {
    const int idx = tid + i * NTHREADS;
    const int g   = idx >> 10;
    const int row = (idx >> 4) & (BN - 1);
    const int c4  = idx & 15;
    *(uint2*)(&smem[bOff + (g * BN + row) * BK + c4 * 4]) = pk_bf16x4(sb[i]);
  }
}

__global__ __launch_bounds__(NTHREADS) void lstm_fused_wmma(
    const float* __restrict__ pre_h, const float* __restrict__ pre_c,
    const float* __restrict__ x,     const float* __restrict__ W,
    const float* __restrict__ U,
    float* __restrict__ outH, float* __restrict__ outC)
{
  extern __shared__ unsigned short smem[];
  // layout: A buf0 | A buf1 | B buf0 | B buf1  (offsets, no LDS pointer arrays)

  const int tid   = threadIdx.x;
  const int lane  = tid & 31;
  const int wid   = tid >> 5;       // 0..15
  const int waveM = wid >> 2;       // 0..3  -> 32-row slab
  const int waveN = wid & 3;        // 0..3  -> 16-col slab

  const int blockN = blockIdx.x * BN;   // unit offset
  const int blockM = blockIdx.y * BM;   // batch offset

  // wave tile: 32(M) x 16(N) per gate -> acc[gate][tm]
  v8f acc[4][2];
  const v8f vzero = {0.f, 0.f, 0.f, 0.f, 0.f, 0.f, 0.f, 0.f};
#pragma unroll
  for (int g = 0; g < 4; ++g)
#pragma unroll
    for (int m = 0; m < 2; ++m)
      acc[g][m] = vzero;

  float4 sa[A_LD];
  float4 sb[B_LD];

  // prologue: stage 0 into buffer 0
  load_stage(x, pre_h, W, U, 0, blockM, blockN, tid, sa, sb);
  store_stage(smem, 0, B_BASE, tid, sa, sb);

  for (int s = 0; s < NSTAGES; ++s) {
    // kick off global loads for the next stage before computing this one
    if (s + 1 < NSTAGES)
      load_stage(x, pre_h, W, U, (s + 1) * BK, blockM, blockN, tid, sa, sb);

    __syncthreads();   // buffer (s&1) fully written, prev-buffer reads done

    const unsigned short* bufA = smem + (s & 1) * A_ELEMS;
    const unsigned short* bufB = smem + B_BASE + (s & 1) * B_ELEMS;

    // ---- compute: two 16x16x32 WMMA K-steps per stage ----
#pragma unroll
    for (int kk = 0; kk < BK; kk += 32) {
      // CDNA5 16-bit A/B fragment layout:
      // lanes 0-15: K = kk+0..7 then kk+16..23 ; lanes 16-31: K = kk+8..15 then kk+24..31
      const int kl = kk + 8 * (lane >> 4);

      BFrag a[2];
#pragma unroll
      for (int tm = 0; tm < 2; ++tm) {
        const unsigned short* p =
            &bufA[(waveM * 32 + tm * 16 + (lane & 15)) * BK + kl];
        a[tm].q[0] = *(const uint4*)(p);
        a[tm].q[1] = *(const uint4*)(p + 16);
      }

#pragma unroll
      for (int g = 0; g < 4; ++g) {
        BFrag b;
        {
          const unsigned short* p =
              &bufB[(g * BN + waveN * 16 + (lane & 15)) * BK + kl];
          b.q[0] = *(const uint4*)(p);
          b.q[1] = *(const uint4*)(p + 16);
        }
#pragma unroll
        for (int tm = 0; tm < 2; ++tm)
          acc[g][tm] = __builtin_amdgcn_wmma_f32_16x16x32_bf16(
              false, a[tm].v, false, b.v,
              (short)0, acc[g][tm], false, false);
      }
    }

    // convert + store next stage into the other buffer (overlaps WMMA drain)
    if (s + 1 < NSTAGES) {
      const int aN = ((s + 1) & 1) * A_ELEMS;
      const int bN = B_BASE + ((s + 1) & 1) * B_ELEMS;
      store_stage(smem, aN, bN, tid, sa, sb);
    }
  }

  // ---- fused LSTM epilogue (gates never leave registers) ----
  // C/D layout: VGPR e, lane L -> M = e + 8*(L/16), N = L%16
  const int r0 = blockM + waveM * 32 + ((lane >> 4) << 3);
  const int c0 = blockN + waveN * 16 + (lane & 15);
#pragma unroll
  for (int tm = 0; tm < 2; ++tm) {
    const v8f gi = acc[0][tm];
    const v8f gf = acc[1][tm];
    const v8f go = acc[2][tm];
    const v8f gn = acc[3][tm];
#pragma unroll
    for (int e = 0; e < 8; ++e) {
      const int row = r0 + tm * 16 + e;
      const size_t idx = (size_t)row * UNITS + c0;
      const float pc = pre_c[idx];
      const float it = fsigmoid(gi[e]);
      const float ft = fsigmoid(gf[e]);
      const float ot = fsigmoid(go[e]);
      const float nt = ftanh(gn[e]);
      const float cc = ft * pc + it * nt;
      const float hh = ot * ftanh(cc);
      outH[idx] = hh;
      outC[idx] = cc;
    }
  }
}

extern "C" void kernel_launch(void* const* d_in, const int* in_sizes, int n_in,
                              void* d_out, int out_size, void* d_ws, size_t ws_size,
                              hipStream_t stream) {
  const float* pre   = (const float*)d_in[0];       // (2, 4096, 2048)
  const float* x     = (const float*)d_in[1];       // (4096, 2048)
  const float* W     = (const float*)d_in[2];       // (4, 2048, 2048)
  const float* U     = (const float*)d_in[3];       // (4, 2048, 2048)
  float* out = (float*)d_out;                       // (2, 4096, 2048)

  const float* pre_h = pre;
  const float* pre_c = pre + (size_t)BATCH * UNITS;
  float* outH = out;
  float* outC = out + (size_t)BATCH * UNITS;

  dim3 grid(UNITS / BN, BATCH / BM);   // (32, 32)
  lstm_fused_wmma<<<grid, NTHREADS, LDS_BYTES, stream>>>(pre_h, pre_c, x, W, U, outH, outC);
}
